// LSTMTagger_44186623541344
// MI455X (gfx1250) — matrix operand order repaired and checked by reading, not compiled, not run on hardware
//
#include <hip/hip_runtime.h>

// ---------------- CDNA5 WMMA types / helpers ----------------
typedef __attribute__((ext_vector_type(16))) __bf16 v16bf;
typedef __attribute__((ext_vector_type(8)))  __bf16 v8bf;
typedef __attribute__((ext_vector_type(8)))  float  v8f;

__device__ __forceinline__ v8f wmma_bf(v16bf a, v16bf b, v8f c) {
  // D = A(16x32) * B(32x16) + C, f32 accumulate
  return __builtin_amdgcn_wmma_f32_16x16x32_bf16(false, a, false, b, (short)0, c,
                                                 false, false);
}

__device__ __forceinline__ v8f zero8() {
  v8f z;
#pragma unroll
  for (int i = 0; i < 8; ++i) z[i] = 0.f;
  return z;
}

__device__ __forceinline__ float sigf(float x) { return 1.f / (1.f + __expf(-x)); }

// A-fragment (16x32 bf16), contiguous k order matching the B packing:
//  lane L holds row M = L%16; lanes 0-15: K in {0..15}; lanes 16-31: K in {16..31}
__device__ __forceinline__ v16bf load_a(const __bf16* base, int rowStride,
                                        int row0, int kt, int lane) {
  int hi8 = ((lane >> 4) & 1) * 16;
  const __bf16* p = base + (size_t)(row0 + (lane & 15)) * rowStride + kt * 32 + hi8;
  v16bf a;
  ((v8bf*)&a)[0] = *(const v8bf*)(p);
  ((v8bf*)&a)[1] = *(const v8bf*)(p + 8);
  return a;
}

// B-fragment (32x16 bf16) from pre-packed weights: tile (nt,kt) is 32 lanes x 16 bf16,
// lane L holds column N = L%16, K = kt*32 + (L<16 ? 0..15 : 16..31), contiguous 32B.
__device__ __forceinline__ v16bf load_b(const __bf16* pk, int Kt, int nt, int kt, int lane) {
  return *(const v16bf*)(pk + (((size_t)nt * Kt + kt) * 32 + lane) * 16);
}

// ---------------- gfx1250 async global->LDS copy (ASYNCcnt-tracked DMA) ----------------
#if defined(__has_builtin)
#if __has_builtin(__builtin_amdgcn_global_load_async_to_lds_b128) && \
    __has_builtin(__builtin_amdgcn_s_wait_asynccnt)
#define HAVE_ASYNC 1
#endif
#endif
#ifndef HAVE_ASYNC
#define HAVE_ASYNC 0
#endif

#if HAVE_ASYNC
typedef int v4i_ __attribute__((ext_vector_type(4)));
typedef __attribute__((address_space(1))) v4i_ g_v4i;   // global (device) int4
typedef __attribute__((address_space(3))) v4i_ l_v4i;   // LDS (shared) int4
// per-lane: LDS[ldsaddr(lane)] <= GLOBAL[gaddr(lane)], 16 bytes, tracked by ASYNCcnt
__device__ __forceinline__ void async_g2l_16B(const void* g, void* l) {
  __builtin_amdgcn_global_load_async_to_lds_b128(
      (g_v4i*)(unsigned long long)(size_t)g,
      (l_v4i*)(unsigned)(size_t)l,   // low 32 bits of a generic LDS ptr = LDS offset
      0, 0);
}
#endif

// ---------------- problem constants ----------------
constexpr int Bb = 32, Tt = 256, Ee = 1024, HDd = 512, H2d = 1024;
constexpr int TAGn = 106, TAGp = 112;
constexpr int ROWS = Bb * Tt;  // 8192, row index = t*B + b

// ---------------- embedding gather (f32 -> bf16, time-major rows) ----------------
__global__ void k_gather(const int* __restrict__ sent, const float* __restrict__ embed,
                         __bf16* __restrict__ xe) {
  int row = blockIdx.x;              // row = t*B + b
  int t = row / Bb, b = row % Bb;
  int tok = sent[b * Tt + t];
  const float* src = embed + (size_t)tok * Ee;
  __bf16* dst = xe + (size_t)row * Ee;
  for (int i = threadIdx.x; i < Ee; i += blockDim.x) dst[i] = (__bf16)src[i];
}

// ---------------- pack f32 weight [K,N] row-major -> bf16 B-fragment tiles ----------------
__global__ void k_pack(const float* __restrict__ W, __bf16* __restrict__ out,
                       int K, int N, int Npad) {
  int id = blockIdx.x * blockDim.x + threadIdx.x;
  int lane = id & 31, tile = id >> 5;
  int Kt = K >> 5, Nt = Npad >> 4;
  if (tile >= Kt * Nt) return;
  int kt = tile % Kt, nt = tile / Kt;
  int n = nt * 16 + (lane & 15);
  int kbase = kt * 32 + ((lane >> 4) << 4);    // lanes 16-31: +16
  __bf16* o = out + (size_t)tile * 512 + lane * 16;
#pragma unroll
  for (int j = 0; j < 16; ++j) {
    float v = (n < N) ? W[(size_t)(kbase + j) * N + n] : 0.f;
    o[j] = (__bf16)v;
  }
}

// ---------------- streaming GEMM: D[M,N](bf16) = A[M,K](bf16) @ Bpk + bias(f32) --------
// Block = 4 waves covering a 64x64 tile footprint: wave w owns row-tile mt=mb*4+w and
// 4 accumulators over nt. B tiles for the block are staged into LDS with async DMA,
// double-buffered in chunks of 4 k-tiles, so each B tile is fetched once per block
// (4x less L2 traffic) and each A fragment feeds 4 WMMAs.
__global__ void __launch_bounds__(128) k_gemm_bias(
    const __bf16* __restrict__ A, const __bf16* __restrict__ Bpk,
    const float* __restrict__ bias, __bf16* __restrict__ D, int M, int K, int N) {
  __shared__ __bf16 bs[2][16 * 512];           // 2 x 16 tiles x 1KB = 32KB
  const int lane = threadIdx.x & 31;
  const int w = threadIdx.x >> 5;              // 0..3
  const int Mb = M >> 6, Kt = K >> 5;
  const int mb = blockIdx.x % Mb, nb = blockIdx.x / Mb;
  const int mt = mb * 4 + w;
  const int NC = Kt >> 2;                      // k-chunks of 4 kt

#if HAVE_ASYNC
  // stage 16 tiles (4 nt x 4 kt) of chunk c into bs[c&1]; 8 async b128 per thread
  auto stage = [&](int c) {
    int q = threadIdx.x >> 3;                  // tile 0..15  (nt_i = q>>2, kt_i = q&3)
    int sub = threadIdx.x & 7;
    int nt_i = q >> 2, kt_i = q & 3;
    const __bf16* src = Bpk + (((size_t)(nb * 4 + nt_i) * Kt) + (c * 4 + kt_i)) * 512;
    __bf16* dst = &bs[c & 1][q * 512];
#pragma unroll
    for (int it = 0; it < 8; ++it) {
      int e = sub * 8 + it * 64;               // halves: 16B per lane per issue
      async_g2l_16B(src + e, dst + e);
    }
  };
  stage(0);
#endif

  v8f acc[4];
#pragma unroll
  for (int j = 0; j < 4; ++j) acc[j] = zero8();

  for (int c = 0; c < NC; ++c) {
#if HAVE_ASYNC
    if (c + 1 < NC) {
      stage(c + 1);
      __builtin_amdgcn_s_wait_asynccnt(8);     // chunk c complete, chunk c+1 in flight
    } else {
      __builtin_amdgcn_s_wait_asynccnt(0);
    }
    __syncthreads();                           // bs[c&1] visible to all waves
#endif
#pragma unroll
    for (int ki = 0; ki < 4; ++ki) {
      int kt = c * 4 + ki;
      v16bf a = load_a(A, K, mt * 16, kt, lane);
#pragma unroll
      for (int j = 0; j < 4; ++j) {
#if HAVE_ASYNC
        v16bf b = *(const v16bf*)(&bs[c & 1][((j * 4 + ki) * 32 + lane) * 16]);
#else
        v16bf b = load_b(Bpk, Kt, nb * 4 + j, kt, lane);
#endif
        acc[j] = wmma_bf(a, b, acc[j]);
      }
    }
#if HAVE_ASYNC
    __syncthreads();                           // all reads done before bs[c&1] restaged
#endif
  }

  int mbase = mt * 16 + ((lane >> 4) & 1) * 8;
#pragma unroll
  for (int j = 0; j < 4; ++j) {
    int n = (nb * 4 + j) * 16 + (lane & 15);
    float bv = bias[n];
#pragma unroll
    for (int r = 0; r < 8; ++r)
      D[(size_t)(mbase + r) * N + n] = (__bf16)(acc[j][r] + bv);
  }
}

// ---------------- sequential bidirectional LSTMP (grid=2: dir, 32 waves/block) --------
// Wave w owns hidden tile nh = [16w, 16w+16); cell state c lives in wave registers.
__global__ void __launch_bounds__(1024)
k_lstmp(const __bf16* __restrict__ Xf, const __bf16* __restrict__ Xb,
        const __bf16* __restrict__ WhhFpk, const __bf16* __restrict__ WhhBpk,
        const __bf16* __restrict__ WprFpk, const __bf16* __restrict__ WprBpk,
        __bf16* __restrict__ y) {
  const int dir = blockIdx.x;
  const __bf16* X   = dir ? Xb : Xf;
  const __bf16* Whh = dir ? WhhBpk : WhhFpk;
  const __bf16* Wpr = dir ? WprBpk : WprFpk;

  __shared__ __bf16 sh[Bb * HDd];    // recurrent h (bf16), 32KB
  __shared__ __bf16 shh[Bb * HDd];   // pre-projection o*tanh(c), 32KB

  const int lane = threadIdx.x & 31, w = threadIdx.x >> 5;
  const int hi8 = ((lane >> 4) & 1) * 8;
  const int nh = w * 16 + (lane & 15);
  const int Kt = HDd / 32;           // 16

  for (int i = threadIdx.x; i < Bb * HDd; i += blockDim.x) sh[i] = (__bf16)0.f;
  float cs[2][8];
#pragma unroll
  for (int m = 0; m < 2; ++m)
#pragma unroll
    for (int r = 0; r < 8; ++r) cs[m][r] = 0.f;
  __syncthreads();

  for (int s = 0; s < Tt; ++s) {
    const int tt = dir ? (Tt - 1 - s) : s;
    // ---- phase 1: gates g = X[tt] + h @ Whh ; update c ; write o*tanh(c) to shh ----
#pragma unroll
    for (int mh = 0; mh < 2; ++mh) {
      v8f ai, af, ag, ao;
#pragma unroll
      for (int r = 0; r < 8; ++r) {
        const __bf16* px = X + (size_t)(tt * Bb + mh * 16 + r + hi8) * (4 * HDd);
        ai[r] = (float)px[nh];
        af[r] = (float)px[HDd + nh];
        ag[r] = (float)px[2 * HDd + nh];
        ao[r] = (float)px[3 * HDd + nh];
      }
      for (int kt = 0; kt < Kt; ++kt) {
        v16bf a = load_a(sh, HDd, mh * 16, kt, lane);
        ai = wmma_bf(a, load_b(Whh, Kt, w, kt, lane), ai);
        af = wmma_bf(a, load_b(Whh, Kt, 32 + w, kt, lane), af);
        ag = wmma_bf(a, load_b(Whh, Kt, 64 + w, kt, lane), ag);
        ao = wmma_bf(a, load_b(Whh, Kt, 96 + w, kt, lane), ao);
      }
#pragma unroll
      for (int r = 0; r < 8; ++r) {
        float c = sigf(af[r]) * cs[mh][r] + sigf(ai[r]) * tanhf(ag[r]);
        cs[mh][r] = c;
        float hh = sigf(ao[r]) * tanhf(c);
        shh[(mh * 16 + r + hi8) * HDd + nh] = (__bf16)hh;
      }
    }
    __syncthreads();
    // ---- phase 2: h = shh @ Wproj ; write to sh and y[tt] ----
#pragma unroll
    for (int mh = 0; mh < 2; ++mh) {
      v8f acc = zero8();
      for (int kt = 0; kt < Kt; ++kt) {
        v16bf a = load_a(shh, HDd, mh * 16, kt, lane);
        acc = wmma_bf(a, load_b(Wpr, Kt, w, kt, lane), acc);
      }
#pragma unroll
      for (int r = 0; r < 8; ++r) {
        int row = mh * 16 + r + hi8;
        __bf16 hv = (__bf16)acc[r];
        sh[row * HDd + nh] = hv;
        y[(size_t)(tt * Bb + row) * H2d + dir * HDd + nh] = hv;
      }
    }
    __syncthreads();
  }
}

// ---------------- sequential LSTMO (grid=1, 32 waves; extra recurrent t-state) --------
__global__ void __launch_bounds__(1024)
k_lstmo(const __bf16* __restrict__ Xo, const __bf16* __restrict__ WhhPk,
        const __bf16* __restrict__ WthPk, const __bf16* __restrict__ WprPk,
        __bf16* __restrict__ z) {
  __shared__ __bf16 sh[Bb * H2d];    // h state, 64KB
  __shared__ __bf16 sts[Bb * H2d];   // t state, 64KB

  const int lane = threadIdx.x & 31, w = threadIdx.x >> 5;
  const int hi8 = ((lane >> 4) & 1) * 8;
  const int Kt = H2d / 32;           // 32

  for (int i = threadIdx.x; i < Bb * H2d; i += blockDim.x) {
    sh[i] = (__bf16)0.f;
    sts[i] = (__bf16)0.f;
  }
  float cs[2][2][8], tn[2][2][8];
#pragma unroll
  for (int m = 0; m < 2; ++m)
#pragma unroll
    for (int s2 = 0; s2 < 2; ++s2)
#pragma unroll
      for (int r = 0; r < 8; ++r) cs[m][s2][r] = 0.f;
  __syncthreads();

  for (int t = 0; t < Tt; ++t) {
    // ---- phase 1: g = Xo[t] + h@Whh + tstate@Wth ; c update ; tn in regs ----
#pragma unroll
    for (int mh = 0; mh < 2; ++mh)
#pragma unroll
      for (int ns = 0; ns < 2; ++ns) {
        int nt = 2 * w + ns;                       // 0..63
        int nh = nt * 16 + (lane & 15);
        v8f ai, af, ag, ao;
#pragma unroll
        for (int r = 0; r < 8; ++r) {
          const __bf16* px = Xo + (size_t)(t * Bb + mh * 16 + r + hi8) * (4 * H2d);
          ai[r] = (float)px[nh];
          af[r] = (float)px[H2d + nh];
          ag[r] = (float)px[2 * H2d + nh];
          ao[r] = (float)px[3 * H2d + nh];
        }
        for (int kt = 0; kt < Kt; ++kt) {
          v16bf a = load_a(sh, H2d, mh * 16, kt, lane);
          ai = wmma_bf(a, load_b(WhhPk, Kt, nt, kt, lane), ai);
          af = wmma_bf(a, load_b(WhhPk, Kt, 64 + nt, kt, lane), af);
          ag = wmma_bf(a, load_b(WhhPk, Kt, 128 + nt, kt, lane), ag);
          ao = wmma_bf(a, load_b(WhhPk, Kt, 192 + nt, kt, lane), ao);
        }
        for (int kt = 0; kt < Kt; ++kt) {
          v16bf a = load_a(sts, H2d, mh * 16, kt, lane);
          ai = wmma_bf(a, load_b(WthPk, Kt, nt, kt, lane), ai);
          af = wmma_bf(a, load_b(WthPk, Kt, 64 + nt, kt, lane), af);
          ag = wmma_bf(a, load_b(WthPk, Kt, 128 + nt, kt, lane), ag);
          ao = wmma_bf(a, load_b(WthPk, Kt, 192 + nt, kt, lane), ao);
        }
#pragma unroll
        for (int r = 0; r < 8; ++r) {
          float c = sigf(af[r]) * cs[mh][ns][r] + sigf(ai[r]) * tanhf(ag[r]);
          cs[mh][ns][r] = c;
          tn[mh][ns][r] = sigf(ao[r]) * tanhf(c);
        }
      }
    __syncthreads();  // all reads of sts done -> safe to overwrite
#pragma unroll
    for (int mh = 0; mh < 2; ++mh)
#pragma unroll
      for (int ns = 0; ns < 2; ++ns) {
        int nh = (2 * w + ns) * 16 + (lane & 15);
#pragma unroll
        for (int r = 0; r < 8; ++r)
          sts[(mh * 16 + r + hi8) * H2d + nh] = (__bf16)tn[mh][ns][r];
      }
    __syncthreads();
    // ---- phase 2: h = tstate @ Wproj ; write sh and z[t] ----
#pragma unroll
    for (int mh = 0; mh < 2; ++mh)
#pragma unroll
      for (int ns = 0; ns < 2; ++ns) {
        int nt = 2 * w + ns;
        int nh = nt * 16 + (lane & 15);
        v8f acc = zero8();
        for (int kt = 0; kt < Kt; ++kt) {
          v16bf a = load_a(sts, H2d, mh * 16, kt, lane);
          acc = wmma_bf(a, load_b(WprPk, Kt, nt, kt, lane), acc);
        }
#pragma unroll
        for (int r = 0; r < 8; ++r) {
          int row = mh * 16 + r + hi8;
          __bf16 hv = (__bf16)acc[r];
          sh[row * H2d + nh] = hv;
          z[(size_t)(t * Bb + row) * H2d + nh] = hv;
        }
      }
    __syncthreads();
  }
}

// ---------------- tag head: logits = z @ Wtag + b ; fused row softmax (TAG=106) --------
__global__ void k_tag(const __bf16* __restrict__ Z, const __bf16* __restrict__ Wpk,
                      const float* __restrict__ bias, float* __restrict__ out) {
  int lane = threadIdx.x & 31, wave = threadIdx.x >> 5;
  int mt = blockIdx.x * (blockDim.x >> 5) + wave;
  if (mt >= ROWS / 16) return;
  const int Kt = H2d / 32;  // 32
  v8f acc[7];
#pragma unroll
  for (int j = 0; j < 7; ++j) acc[j] = zero8();
  for (int kt = 0; kt < Kt; ++kt) {
    v16bf a = load_a(Z, H2d, mt * 16, kt, lane);
#pragma unroll
    for (int j = 0; j < 7; ++j)
      acc[j] = wmma_bf(a, load_b(Wpk, Kt, j, kt, lane), acc[j]);
  }
  int ncol = lane & 15;
  int hi8 = ((lane >> 4) & 1) * 8;
#pragma unroll
  for (int r = 0; r < 8; ++r) {
    float l[7];
    float mx = -3.0e38f;
#pragma unroll
    for (int j = 0; j < 7; ++j) {
      int n = j * 16 + ncol;
      l[j] = (n < TAGn) ? (acc[j][r] + bias[n]) : -3.0e38f;
      mx = fmaxf(mx, l[j]);
    }
#pragma unroll
    for (int off = 8; off >= 1; off >>= 1) mx = fmaxf(mx, __shfl_xor(mx, off, 32));
    float e[7], s = 0.f;
#pragma unroll
    for (int j = 0; j < 7; ++j) { e[j] = __expf(l[j] - mx); s += e[j]; }
#pragma unroll
    for (int off = 8; off >= 1; off >>= 1) s += __shfl_xor(s, off, 32);
    float inv = 1.f / s;
    int g = mt * 16 + r + hi8;       // g = t*B + b
    int tb = g >> 5, bb = g & 31;
    float* po = out + ((size_t)bb * Tt + tb) * TAGn;
#pragma unroll
    for (int j = 0; j < 7; ++j) {
      int n = j * 16 + ncol;
      if (n < TAGn) po[n] = e[j] * inv;
    }
  }
}

// ---------------- host-side orchestration ----------------
extern "C" void kernel_launch(void* const* d_in, const int* in_sizes, int n_in,
                              void* d_out, int out_size, void* d_ws, size_t ws_size,
                              hipStream_t stream) {
  (void)in_sizes; (void)n_in; (void)out_size; (void)ws_size;
  const int*   sent  = (const int*)  d_in[0];
  const float* embed = (const float*)d_in[1];
  const float* WihF  = (const float*)d_in[2];
  const float* WhhF  = (const float*)d_in[3];
  const float* bF    = (const float*)d_in[4];
  const float* WprF  = (const float*)d_in[5];
  const float* WihB  = (const float*)d_in[6];
  const float* WhhB  = (const float*)d_in[7];
  const float* bB    = (const float*)d_in[8];
  const float* WprB  = (const float*)d_in[9];
  const float* WihO  = (const float*)d_in[10];
  const float* WhhO  = (const float*)d_in[11];
  const float* WthO  = (const float*)d_in[12];
  const float* bO    = (const float*)d_in[13];
  const float* WprO  = (const float*)d_in[14];
  const float* Wtag  = (const float*)d_in[15];
  const float* btag  = (const float*)d_in[16];
  float* out = (float*)d_out;

  char* ws = (char*)d_ws;
  size_t off = 0;
  auto take = [&](size_t elems) -> __bf16* {
    off = (off + 255) & ~(size_t)255;
    __bf16* p = (__bf16*)(ws + off);
    off += elems * sizeof(__bf16);
    return p;
  };
  __bf16* xe    = take((size_t)ROWS * Ee);          // 16 MB
  __bf16* pWihF = take((size_t)Ee  * 4 * HDd);
  __bf16* pWihB = take((size_t)Ee  * 4 * HDd);
  __bf16* pWhhF = take((size_t)HDd * 4 * HDd);
  __bf16* pWhhB = take((size_t)HDd * 4 * HDd);
  __bf16* pWprF = take((size_t)HDd * HDd);
  __bf16* pWprB = take((size_t)HDd * HDd);
  __bf16* pWihO = take((size_t)H2d * 4 * H2d);
  __bf16* pWhhO = take((size_t)H2d * 4 * H2d);
  __bf16* pWthO = take((size_t)H2d * 4 * H2d);
  __bf16* pWprO = take((size_t)H2d * H2d);
  __bf16* pWtag = take((size_t)H2d * TAGp);
  __bf16* Xf    = take((size_t)ROWS * 4 * HDd);     // 32 MB
  __bf16* Xb    = take((size_t)ROWS * 4 * HDd);     // 32 MB
  __bf16* ybuf  = take((size_t)ROWS * H2d);         // 16 MB
  __bf16* Xo    = take((size_t)ROWS * 4 * H2d);     // 64 MB
  __bf16* zbuf  = take((size_t)ROWS * H2d);         // 16 MB

  // 1) gather embeddings, convert bf16, time-major rows (t*B+b)
  k_gather<<<ROWS, 256, 0, stream>>>(sent, embed, xe);

  // 2) pack all weights into WMMA B-fragment layout
  auto pack = [&](const float* W, __bf16* dst, int K, int N, int Npad) {
    int tiles = (K / 32) * (Npad / 16);
    k_pack<<<(tiles * 32 + 255) / 256, 256, 0, stream>>>(W, dst, K, N, Npad);
  };
  pack(WihF, pWihF, Ee,  4 * HDd, 4 * HDd);
  pack(WihB, pWihB, Ee,  4 * HDd, 4 * HDd);
  pack(WhhF, pWhhF, HDd, 4 * HDd, 4 * HDd);
  pack(WhhB, pWhhB, HDd, 4 * HDd, 4 * HDd);
  pack(WprF, pWprF, HDd, HDd, HDd);
  pack(WprB, pWprB, HDd, HDd, HDd);
  pack(WihO, pWihO, H2d, 4 * H2d, 4 * H2d);
  pack(WhhO, pWhhO, H2d, 4 * H2d, 4 * H2d);
  pack(WthO, pWthO, H2d, 4 * H2d, 4 * H2d);
  pack(WprO, pWprO, H2d, H2d, H2d);
  pack(Wtag, pWtag, H2d, TAGn, TAGp);

  // 3) time-parallel input GEMMs (64x64 block footprint, async-staged B)
  auto gemm = [&](const __bf16* A, const __bf16* Bp, const float* bias, __bf16* D,
                  int M, int K, int N) {
    int blocks = (M / 64) * (N / 64);
    k_gemm_bias<<<blocks, 128, 0, stream>>>(A, Bp, bias, D, M, K, N);
  };
  gemm(xe, pWihF, bF, Xf, ROWS, Ee, 4 * HDd);
  gemm(xe, pWihB, bB, Xb, ROWS, Ee, 4 * HDd);

  // 4) sequential bidirectional LSTMP (fwd & bwd run concurrently, grid=2)
  k_lstmp<<<2, 1024, 0, stream>>>(Xf, Xb, pWhhF, pWhhB, pWprF, pWprB, ybuf);

  // 5) time-parallel input GEMM for LSTMO
  gemm(ybuf, pWihO, bO, Xo, ROWS, H2d, 4 * H2d);

  // 6) sequential LSTMO
  k_lstmo<<<1, 1024, 0, stream>>>(Xo, pWhhO, pWthO, pWprO, zbuf);

  // 7) tag head + softmax
  k_tag<<<(ROWS / 16) / 4, 128, 0, stream>>>(zbuf, pWtag, btag, out);
}
